// MNISTSort2Net_79319456022950
// MI455X (gfx1250) — matrix operand order repaired and checked by reading, not compiled, not run on hardware
//
#include <hip/hip_runtime.h>
#include <hip/hip_bf16.h>

// ---------------------------------------------------------------------------
// MNISTSort2Net forward for MI455X (gfx1250, wave32, WMMA + TDM/async)
//   conv1(1->32,5x5)+pool (VALU, ~7GF) ->
//   conv2(32->64,5x5)+pool as LDS-staged implicit-GEMM bf16 WMMA (~54GF)
//     with TDM tensor_load_to_lds input staging and async LDS->global store ->
//   fc1(1024x1024) bf16 WMMA + relu (~17GF) -> fc2(1024x10) -> softmax ->
//   categorical sampling -> masked histograms -> scalar MSE.
// ---------------------------------------------------------------------------

typedef __attribute__((ext_vector_type(16))) __bf16 v16bf;
typedef __attribute__((ext_vector_type(8)))  float  v8f;
typedef __attribute__((ext_vector_type(4)))  unsigned int v4u;
typedef __attribute__((ext_vector_type(8)))  int    v8i_t;
typedef __attribute__((ext_vector_type(4)))  int    v4i_t;

#define AS1 __attribute__((address_space(1)))
#define AS3 __attribute__((address_space(3)))

__device__ __forceinline__ unsigned short f32_to_bf16_rne(float f) {
    unsigned int u = __float_as_uint(f);
    unsigned int r = u + 0x7FFFu + ((u >> 16) & 1u);   // round to nearest even
    return (unsigned short)(r >> 16);
}

// ------------------------------ conv1 + pool --------------------------------
// imgs [B,1,28,28] f32 -> act1 [B,32,12,12] f32
__global__ __launch_bounds__(256) void conv1_pool_kernel(
    const float* __restrict__ imgs, const float* __restrict__ w,
    const float* __restrict__ bias, float* __restrict__ act1, int B) {
    int idx = blockIdx.x * blockDim.x + threadIdx.x;
    int total = B * 32 * 12 * 12;
    if (idx >= total) return;
    int ow = idx % 12;
    int oh = (idx / 12) % 12;
    int oc = (idx / 144) % 32;
    int b  = idx / (144 * 32);
    const float* im = imgs + b * 784;
    const float* wk = w + oc * 25;
    float bi = bias[oc];
    float mx = -3.4e38f;
    #pragma unroll
    for (int dy = 0; dy < 2; ++dy) {
        #pragma unroll
        for (int dx = 0; dx < 2; ++dx) {
            int ch = oh * 2 + dy, cw = ow * 2 + dx;   // conv output coord (0..23)
            float acc = bi;
            #pragma unroll
            for (int ky = 0; ky < 5; ++ky)
                #pragma unroll
                for (int kx = 0; kx < 5; ++kx)
                    acc = fmaf(im[(ch + ky) * 28 + (cw + kx)], wk[ky * 5 + kx], acc);
            mx = fmaxf(mx, acc);
        }
    }
    act1[idx] = mx;
}

// ---------------- weight converts (f32 -> bf16, fc1 transposed) -------------
__global__ __launch_bounds__(256) void w1_transpose_kernel(
    const float* __restrict__ w, unsigned short* __restrict__ wt) {
    int idx = blockIdx.x * blockDim.x + threadIdx.x;   // idx = n*1024 + k
    if (idx >= 1024 * 1024) return;
    int k = idx & 1023;
    int n = idx >> 10;
    wt[idx] = f32_to_bf16_rne(w[k * 1024 + n]);
}

// conv2_w [64][32][5][5] f32 is already [n][k=ic*25+ky*5+kx] row-major
__global__ __launch_bounds__(256) void w2_convert_kernel(
    const float* __restrict__ w, unsigned short* __restrict__ wh) {
    int i = blockIdx.x * blockDim.x + threadIdx.x;
    if (i < 64 * 800) wh[i] = f32_to_bf16_rne(w[i]);
}

// ------------------- conv2 + pool: implicit-GEMM bf16 WMMA ------------------
// Per image b: M=64 conv positions (8x8), N=64 out channels, K=800=25*32.
// LDS layout (dynamic, 122880 B):
//   [0      .. 102400) : 64x800 bf16 im2col tile (aliased as 64x64 f32 conv map)
//   [102400 .. 120832) : raw act1[b] slab (32x12x12 f32), TDM-staged
//   [120832 .. 122880) : pooled 1024 bf16 outputs, async-stored to global
__global__ __launch_bounds__(256) void conv2_pool_wmma_kernel(
    const float* __restrict__ act1,            // [B,32,12,12]
    const unsigned short* __restrict__ w2h,    // [64][800] bf16
    const float* __restrict__ bias,
    unsigned short* __restrict__ act2h,        // [B,1024] bf16 (flatten oc*16+oh*4+ow)
    int B) {
    extern __shared__ char smem[];
    unsigned short* im   = (unsigned short*)smem;             // [64][800] bf16
    float*          conv = (float*)smem;                      // [64][64] f32 (alias)
    float*          raw  = (float*)(smem + 102400);           // [4608] f32
    unsigned short* pool = (unsigned short*)(smem + 120832);  // [1024] bf16
    const int b   = blockIdx.x;
    const int tid = threadIdx.x;
    const float* in = act1 + (size_t)b * 4608;

    // ---- stage raw act1[b] (18432 B contiguous) into LDS
#if __has_builtin(__builtin_amdgcn_tensor_load_to_lds)
    if (tid < 32) {                // wave 0 issues one TDM descriptor
        unsigned long long ga = (unsigned long long)(const void*)in;
        // D# group0: count=1 | lds_addr | global_addr[56:0] | type=2
        v4u g0 = { 1u,
                   102400u,
                   (unsigned int)ga,
                   (unsigned int)(((ga >> 32) & 0x1FFFFFFu) | (2u << 30)) };
        // D# group1: data_size=4B(code 2); tensor_dim0=4608 @[79:48];
        //            tensor_dim1=1 @[111:80]; tile_dim0=4608 @[127:112];
        //            tile_dim1=1 @[143:128]; tensor_dim0_stride=4608 @[207:160]
        v8i_t g1 = { 0x20000,
                     (int)(4608u << 16),
                     (int)(1u << 16),
                     (int)(4608u << 16),
                     1,
                     4608,
                     0,
                     0 };
        v4i_t g2 = { 0, 0, 0, 0 };     // 2D tensor: groups 2/3 unused
        v4i_t g3 = { 0, 0, 0, 0 };
        v8i_t g4 = { 0, 0, 0, 0, 0, 0, 0, 0 };
        __builtin_amdgcn_tensor_load_to_lds(g0, g1, g2, g3, g4, 0);
        __builtin_amdgcn_s_wait_tensorcnt(0);
    }
#else
    for (int i = tid; i < 4608; i += 256) raw[i] = in[i];
#endif
    __syncthreads();

    // ---- im2col from LDS raw: m = ch*8+cw, k = ic*25 + ky*5 + kx
    for (int i = tid; i < 64 * 800; i += 256) {
        int m = i / 800, k = i - m * 800;
        int ch = m >> 3, cw = m & 7;
        int ic = k / 25, r = k - ic * 25;
        int ky = r / 5,  kx = r - ky * 5;
        im[i] = f32_to_bf16_rne(raw[ic * 144 + (ch + ky) * 12 + (cw + kx)]);
    }
    __syncthreads();

    const int lane = tid & 31, wave = tid >> 5;
    const int row = lane & 15, hi = lane >> 4;     // half-wave selector
    const int tn  = wave & 3;                      // N-tile (shared by both tiles)
    const int tm1 = wave >> 2;                     // M-tiles: {0,1} and {2,3}
    const int tm2 = tm1 + 2;
    const unsigned short* pB  = w2h + (size_t)(tn * 16 + row) * 800 + hi * 8;
    const unsigned short* pA1 = im  + (size_t)(tm1 * 16 + row) * 800 + hi * 8;
    const unsigned short* pA2 = im  + (size_t)(tm2 * 16 + row) * 800 + hi * 8;

    v8f acc0 = {}, acc1 = {};
    for (int k0 = 0; k0 < 800; k0 += 32) {
        union { uint4 u[2]; v16bf v; } bf2, a1, a2;
        bf2.u[0] = *(const uint4*)(pB + k0);
        bf2.u[1] = *(const uint4*)(pB + k0 + 16);
        a1.u[0] = *(const uint4*)(pA1 + k0);
        a1.u[1] = *(const uint4*)(pA1 + k0 + 16);
        a2.u[0] = *(const uint4*)(pA2 + k0);
        a2.u[1] = *(const uint4*)(pA2 + k0 + 16);
        acc0 = __builtin_amdgcn_wmma_f32_16x16x32_bf16(
            false, a1.v, false, bf2.v, (short)0, acc0, false, false);
        acc1 = __builtin_amdgcn_wmma_f32_16x16x32_bf16(
            false, a2.v, false, bf2.v, (short)0, acc1, false, false);
    }
    __syncthreads();        // im2col no longer needed; reuse LDS for conv map

    const int n  = tn * 16 + row;
    const float bn = bias[n];
    #pragma unroll
    for (int r = 0; r < 8; ++r) {
        conv[(tm1 * 16 + hi * 8 + r) * 64 + n] = acc0[r] + bn;
        conv[(tm2 * 16 + hi * 8 + r) * 64 + n] = acc1[r] + bn;
    }
    __syncthreads();

    // ---- fused 2x2 maxpool + bf16 convert into LDS staging: j = oc*16+oh*4+ow
    #pragma unroll
    for (int i = 0; i < 4; ++i) {
        int j  = tid + i * 256;                    // 0..1023
        int oc = j >> 4;
        int oh = (j >> 2) & 3, ow = j & 3;
        int ch = oh * 2, cw = ow * 2;
        float v00 = conv[(ch * 8 + cw) * 64 + oc];
        float v01 = conv[(ch * 8 + cw + 1) * 64 + oc];
        float v10 = conv[((ch + 1) * 8 + cw) * 64 + oc];
        float v11 = conv[((ch + 1) * 8 + cw + 1) * 64 + oc];
        pool[j] = f32_to_bf16_rne(fmaxf(fmaxf(v00, v01), fmaxf(v10, v11)));
    }
    __syncthreads();

    // ---- drain pooled tile (2048 B) to global via async LDS->memory path
#if __has_builtin(__builtin_amdgcn_global_store_async_from_lds_b128)
    if (tid < 128) {
        unsigned short* dst = act2h + (size_t)b * 1024 + tid * 8;  // 16 B/lane
        __builtin_amdgcn_global_store_async_from_lds_b128(
            (AS1 v4i_t*)(unsigned long long)(void*)dst,
            (AS3 v4i_t*)(unsigned long long)(void*)(pool + tid * 8),
            0, 0);
#if __has_builtin(__builtin_amdgcn_s_wait_asynccnt)
        __builtin_amdgcn_s_wait_asynccnt(0);
#else
        asm volatile("s_wait_asynccnt 0" ::: "memory");
#endif
    }
#else
    for (int i = tid; i < 1024; i += 256)
        act2h[(size_t)b * 1024 + i] = pool[i];
#endif
}

// --------------------------- fc1: WMMA bf16 GEMM ----------------------------
// out[m][n] = relu( sum_k A[m][k] * Wt[n][k] + bias[n] ),  M=B, N=K=1024
__global__ __launch_bounds__(256) void fc1_wmma_kernel(
    const unsigned short* __restrict__ A,    // [M,1024] bf16 row-major
    const unsigned short* __restrict__ Bt,   // [1024,1024] bf16, Bt[n][k]
    const float* __restrict__ bias, float* __restrict__ out) {
    const int lane = threadIdx.x & 31;
    const int wave = threadIdx.x >> 5;
    const int tile = blockIdx.x * 8 + wave;
    const int tn = tile & 63;                // 1024/16 = 64 N-tiles
    const int tm = tile >> 6;
    const int m0 = tm * 16, n0 = tn * 16;
    const int row = lane & 15;
    const int hi  = lane >> 4;
    const unsigned short* pA = A  + (size_t)(m0 + row) * 1024 + hi * 8;
    const unsigned short* pB = Bt + (size_t)(n0 + row) * 1024 + hi * 8;

    v8f acc = {};
    for (int k0 = 0; k0 < 1024; k0 += 32) {
        __builtin_prefetch((const void*)(pA + k0 + 256), 0, 1);
        __builtin_prefetch((const void*)(pB + k0 + 256), 0, 1);
        union { uint4 u[2]; v16bf v; } af, bf2;
        af.u[0] = *(const uint4*)(pA + k0);
        af.u[1] = *(const uint4*)(pA + k0 + 16);
        bf2.u[0] = *(const uint4*)(pB + k0);
        bf2.u[1] = *(const uint4*)(pB + k0 + 16);
        acc = __builtin_amdgcn_wmma_f32_16x16x32_bf16(
            false, af.v, false, bf2.v, (short)0, acc, false, false);
    }

    const int n = n0 + row;
    const float bn = bias[n];
    #pragma unroll
    for (int r = 0; r < 8; ++r) {
        int m = m0 + hi * 8 + r;
        out[(size_t)m * 1024 + n] = fmaxf(acc[r] + bn, 0.0f);
    }
}

// --------------------------- fc2 logits + softmax ---------------------------
__global__ __launch_bounds__(256) void fc2_logits_kernel(
    const float* __restrict__ x, const float* __restrict__ w,
    const float* __restrict__ bias, float* __restrict__ logits, int B) {
    int idx = blockIdx.x * blockDim.x + threadIdx.x;   // idx = b*10 + c
    if (idx >= B * 10) return;
    int c = idx % 10;
    int b = idx / 10;
    const float* xr = x + (size_t)b * 1024;
    float acc = bias[c];
    for (int k = 0; k < 1024; ++k)
        acc = fmaf(xr[k], w[k * 10 + c], acc);
    logits[idx] = acc;
}

__global__ __launch_bounds__(256) void softmax_kernel(
    const float* __restrict__ logits, float* __restrict__ distrs, int B) {
    int b = blockIdx.x * blockDim.x + threadIdx.x;
    if (b >= B) return;
    float v[10], mx = -3.4e38f;
    #pragma unroll
    for (int c = 0; c < 10; ++c) { v[c] = logits[b * 10 + c]; mx = fmaxf(mx, v[c]); }
    float s = 0.f;
    #pragma unroll
    for (int c = 0; c < 10; ++c) { v[c] = __expf(v[c] - mx); s += v[c]; }
    float inv = 1.0f / s;
    #pragma unroll
    for (int c = 0; c < 10; ++c) distrs[b * 10 + c] = v[c] * inv;
}

// --------------------- categorical sampling + histograms --------------------
__device__ __forceinline__ float hash_uniform(unsigned int b, unsigned int s,
                                              unsigned int c) {
    unsigned int h = b * 2654435761u ^ (s + 0x9E3779B9u) ^ (c * 0x85EBCA6Bu + 0xC2B2AE35u);
    h ^= h >> 16; h *= 0x7FEB352Du;
    h ^= h >> 15; h *= 0x846CA68Bu;
    h ^= h >> 16;
    return (float)(h >> 8) * (1.0f / 16777216.0f);
}

__global__ __launch_bounds__(256) void sample_hist_kernel(
    const float* __restrict__ da, const float* __restrict__ db,
    const int* __restrict__ y, float* __restrict__ pa, float* __restrict__ pb,
    int S) {
    int b = blockIdx.x;
    __shared__ float cdfa[10], cdfb[10];
    __shared__ unsigned int histA[10], histB[10], total;
    if (threadIdx.x == 0) {
        float ca = 0.f, cb = 0.f;
        for (int c = 0; c < 10; ++c) {
            ca += da[b * 10 + c]; cdfa[c] = ca;
            cb += db[b * 10 + c]; cdfb[c] = cb;
        }
        total = 0u;
    }
    if (threadIdx.x < 10) { histA[threadIdx.x] = 0u; histB[threadIdx.x] = 0u; }
    __syncthreads();
    int yb = y[b];
    for (int s = threadIdx.x; s < S; s += blockDim.x) {
        float u1 = hash_uniform(b, s, 0) * cdfa[9];
        float u2 = hash_uniform(b, s, 1) * cdfb[9];
        int as = 9, bs = 9;
        #pragma unroll
        for (int c = 9; c >= 0; --c) if (u1 < cdfa[c]) as = c;
        #pragma unroll
        for (int c = 9; c >= 0; --c) if (u2 < cdfb[c]) bs = c;
        if (bs == yb && as >= bs) {
            atomicAdd(&histA[as], 1u);
            atomicAdd(&histB[bs], 1u);
            atomicAdd(&total, 1u);
        }
    }
    __syncthreads();
    if (threadIdx.x < 10) {
        float t = (float)total;
        float denom = fmaxf(t, 1.0f);
        float nz = (t > 0.f) ? 1.0f : 0.0f;
        pa[b * 10 + threadIdx.x] = nz * (float)histA[threadIdx.x] / denom;
        pb[b * 10 + threadIdx.x] = nz * (float)histB[threadIdx.x] / denom;
    }
}

// ------------------------------- final MSE ----------------------------------
__global__ void zero_out_kernel(float* out) { out[0] = 0.0f; }

__global__ __launch_bounds__(256) void mse_kernel(
    const float* __restrict__ da, const float* __restrict__ db,
    const float* __restrict__ pa, const float* __restrict__ pb,
    float* __restrict__ out, int n /* = B*10 */) {
    int i = blockIdx.x * blockDim.x + threadIdx.x;
    float v = 0.f;
    if (i < 2 * n) {
        float d = (i < n) ? (da[i] - pa[i]) : (db[i - n] - pb[i - n]);
        v = d * d / (2.0f * (float)n);
    }
    #pragma unroll
    for (int off = 16; off > 0; off >>= 1)
        v += __shfl_down(v, off, 32);
    if ((threadIdx.x & 31) == 0 && v != 0.f) atomicAdd(out, v);
}

// ---------------------------------------------------------------------------
extern "C" void kernel_launch(void* const* d_in, const int* in_sizes, int n_in,
                              void* d_out, int out_size, void* d_ws, size_t ws_size,
                              hipStream_t stream) {
    const float* a_imgs  = (const float*)d_in[0];
    const float* b_imgs  = (const float*)d_in[1];
    const float* conv1_w = (const float*)d_in[2];
    const float* conv1_b = (const float*)d_in[3];
    const float* conv2_w = (const float*)d_in[4];
    const float* conv2_b = (const float*)d_in[5];
    const float* fc1_w   = (const float*)d_in[6];
    const float* fc1_b   = (const float*)d_in[7];
    const float* fc2_w   = (const float*)d_in[8];
    const float* fc2_b   = (const float*)d_in[9];
    const int*   y       = (const int*)d_in[10];
    const int    S       = 1024;            // n_samples (fixed by setup_inputs)

    const int B = in_sizes[0] / 784;        // 4096

    // workspace carve-up
    char* ws = (char*)d_ws;
    float*          act1   = (float*)ws;            ws += (size_t)B * 32 * 144 * sizeof(float);
    unsigned short* act2h  = (unsigned short*)ws;   ws += (size_t)B * 1024 * sizeof(unsigned short);
    unsigned short* wt1    = (unsigned short*)ws;   ws += (size_t)1024 * 1024 * sizeof(unsigned short);
    unsigned short* w2h    = (unsigned short*)ws;   ws += (size_t)64 * 800 * sizeof(unsigned short);
    float*          fc1out = (float*)ws;            ws += (size_t)B * 1024 * sizeof(float);
    float*          logits = (float*)ws;            ws += (size_t)B * 10 * sizeof(float);
    float*          distrA = (float*)ws;            ws += (size_t)B * 10 * sizeof(float);
    float*          distrB = (float*)ws;            ws += (size_t)B * 10 * sizeof(float);
    float*          predA  = (float*)ws;            ws += (size_t)B * 10 * sizeof(float);
    float*          predB  = (float*)ws;            ws += (size_t)B * 10 * sizeof(float);

    // weight conversions (shared by both batches)
    w1_transpose_kernel<<<(1024 * 1024 + 255) / 256, 256, 0, stream>>>(fc1_w, wt1);
    w2_convert_kernel<<<(64 * 800 + 255) / 256, 256, 0, stream>>>(conv2_w, w2h);

    const size_t conv2_lds = 122880;   // im2col 102400 + raw 18432 + pool 2048

    const float* imgs[2]  = {a_imgs, b_imgs};
    float*       distr[2] = {distrA, distrB};
    for (int p = 0; p < 2; ++p) {
        conv1_pool_kernel<<<((size_t)B * 32 * 144 + 255) / 256, 256, 0, stream>>>(
            imgs[p], conv1_w, conv1_b, act1, B);
        conv2_pool_wmma_kernel<<<B, 256, conv2_lds, stream>>>(
            act1, w2h, conv2_b, act2h, B);
        fc1_wmma_kernel<<<(B / 16) * 64 / 8, 256, 0, stream>>>(
            act2h, wt1, fc1_b, fc1out);
        fc2_logits_kernel<<<((size_t)B * 10 + 255) / 256, 256, 0, stream>>>(
            fc1out, fc2_w, fc2_b, logits, B);
        softmax_kernel<<<(B + 255) / 256, 256, 0, stream>>>(logits, distr[p], B);
    }

    sample_hist_kernel<<<B, 256, 0, stream>>>(distrA, distrB, y, predA, predB, S);

    zero_out_kernel<<<1, 1, 0, stream>>>((float*)d_out);
    mse_kernel<<<((size_t)2 * B * 10 + 255) / 256, 256, 0, stream>>>(
        distrA, distrB, predA, predB, (float*)d_out, B * 10);
}
